// RNN_80307298500754
// MI455X (gfx1250) — matrix operand order repaired and checked by reading, not compiled, not run on hardware
//
#include <hip/hip_runtime.h>
#include <hip/hip_bf16.h>

typedef __attribute__((ext_vector_type(16))) _Float16 v16h;
typedef __attribute__((ext_vector_type(8)))  _Float16 v8h;
typedef __attribute__((ext_vector_type(4)))  _Float16 v4h;
typedef __attribute__((ext_vector_type(8)))  float    v8f;
typedef __attribute__((ext_vector_type(4)))  int      v4i;

#define Bc  128
#define Tc  1024
#define Vc  128
#define Ec  64
#define Hc  64
#define XHc 128   // E + H

__device__ __forceinline__ float rcp_(float x) { return __builtin_amdgcn_rcpf(x); }
__device__ __forceinline__ float sigmoidf_(float x) {
  return rcp_(1.0f + __expf(-x));                 // v_exp + v_add + v_rcp
}
__device__ __forceinline__ float tanhf_(float x) {
  return 1.0f - 2.0f * rcp_(__expf(2.0f * x) + 1.0f);
}

// ---- A-fragment (16x32 f16, MxK) from row-major f16 LDS (16B-aligned rows).
// ISA layout: lane<16 -> M=lane, halves 0..7 = K+0..7, halves 8..15 = K+16..23;
// lanes 16..31 -> same M, K offset +8.  Two aligned 16B loads per lane.
__device__ __forceinline__ v16h afrag_lds(const _Float16* m, int ld, int k0, int lane) {
  const _Float16* p = m + (lane & 15) * ld + k0 + ((lane >> 4) << 3);
  v8h lo = *(const v8h*)(p);
  v8h hi = *(const v8h*)(p + 16);
  return __builtin_shufflevector(lo, hi, 0, 1, 2, 3, 4, 5, 6, 7,
                                         8, 9, 10, 11, 12, 13, 14, 15);
}

// ---- B-fragment (32x16 f16, KxN) for X @ W^T : B[k][n] = W[n][k].
// lane = N (lanes 0-15), K+0..7 and K+16..23 per lane; lanes 16-31 hold K+8.
__device__ __forceinline__ v16h bfrag_f32(const float* w, int ld, int row0, int k0, int lane) {
  const float* p = w + (row0 + (lane & 15)) * ld + k0 + ((lane >> 4) << 3);
  v16h f;
#pragma unroll
  for (int i = 0; i < 16; ++i) f[i] = (_Float16)p[i + (i & 8)];
  return f;
}

__device__ __forceinline__ v8f wmma16(v16h a, v16h b, v8f c) {
  return __builtin_amdgcn_wmma_f32_16x16x32_f16(false, a, false, b, (short)0, c, false, false);
}

// ---------------------------------------------------------------------------
// Kernel 1: sequential GRU scan. One workgroup = 16 batch rows, 8 waves.
// Waves 0-3: z-gate (kept in regs) then hbar; waves 4-7: r-gate.
// All inputs (idx slice, wte, weights) resident in LDS/registers for the scan.
// ---------------------------------------------------------------------------
__global__ __launch_bounds__(256, 1) void gru_scan_kernel(
    const int* __restrict__ idx, const float* __restrict__ start,
    const float* __restrict__ wte,
    const float* __restrict__ Wz, const float* __restrict__ bz,
    const float* __restrict__ Wr, const float* __restrict__ br,
    const float* __restrict__ Wh, const float* __restrict__ bh,
    _Float16* __restrict__ hidden)   // [B, T, H] f16
{
  __shared__ __align__(16) int      idx_lds[16 * Tc];   // 64 KB: token ids
  __shared__ __align__(16) _Float16 wte_lds[Vc * Ec];   // 16 KB: embeddings f16
  __shared__ __align__(16) _Float16 xh[16 * XHc];       //  4 KB: [x_t | h] / [x_t | r*h]
  __shared__ __align__(16) float    h_lds[16 * Hc];     //  4 KB: hidden state f32
  __shared__ __align__(16) float    r_lds[16 * Hc];     //  4 KB: sigmoid(r)

  const int tid  = threadIdx.x;
  const int lane = tid & 31;
  const int wave = tid >> 5;
  const int b0   = blockIdx.x * 16;
  const int nt   = wave & 3;                    // N-tile of H (4 x 16)
  const int mrow = (lane >> 4) << 3;            // C-layout row base
  const int ncol = nt * 16 + (lane & 15);       // C-layout column (H index)

  // Contiguous 64KB copy of this block's idx rows: idx[b0*T .. (b0+16)*T).
  {
    const v4i* src = (const v4i*)(idx + (size_t)b0 * Tc);
    v4i* dst = (v4i*)idx_lds;
    for (int i = tid; i < (16 * Tc) / 4; i += 256) dst[i] = src[i];
  }
  for (int i = tid; i < Vc * Ec; i += 256) wte_lds[i] = (_Float16)wte[i];
  for (int i = tid; i < 16 * Hc; i += 256) h_lds[i] = start[i & (Hc - 1)];

  // Weight B-fragments live in registers for the whole scan.
  v16h wza[4] = {}, wha[4] = {}, wra[4] = {};
  if (wave < 4) {
#pragma unroll
    for (int kb = 0; kb < 4; ++kb) {
      wza[kb] = bfrag_f32(Wz, XHc, nt * 16, kb * 32, lane);
      wha[kb] = bfrag_f32(Wh, XHc, nt * 16, kb * 32, lane);
    }
  } else {
#pragma unroll
    for (int kb = 0; kb < 4; ++kb)
      wra[kb] = bfrag_f32(Wr, XHc, nt * 16, kb * 32, lane);
  }
  const float bzv = bz[ncol];
  const float brv = br[ncol];
  const float bhv = bh[ncol];

  // Per-thread staging coordinates.
  const int srow = tid >> 4;            // 0..15
  const int sc8  = (tid & 15) << 3;     // 0,8,..,120 (xh build: 8 halves)
  const int sc4  = (tid & 15) << 2;     // 0,4,..,60  (H-wide ops: 4 elems)
  __syncthreads();

  for (int t = 0; t < Tc; ++t) {
    // Build xh = [x_t | h] : one aligned 16B store per thread.
    {
      v8h v;
      if (sc8 < Ec) {
        int tok = idx_lds[srow * Tc + t];
        v = *(const v8h*)(wte_lds + tok * Ec + sc8);
      } else {
        const float* hp = h_lds + srow * Hc + (sc8 - Ec);
#pragma unroll
        for (int j = 0; j < 8; ++j) v[j] = (_Float16)hp[j];
      }
      *(v8h*)(xh + srow * XHc + sc8) = v;
    }
    __syncthreads();

    // Phase A: z (waves 0-3, kept in regs) and r (waves 4-7 -> LDS).
    v8f zacc;
#pragma unroll
    for (int j = 0; j < 8; ++j) zacc[j] = bzv;
    if (wave < 4) {
#pragma unroll
      for (int kb = 0; kb < 4; ++kb)
        zacc = wmma16(afrag_lds(xh, XHc, kb * 32, lane), wza[kb], zacc);
#pragma unroll
      for (int j = 0; j < 8; ++j) zacc[j] = sigmoidf_(zacc[j]);
    } else {
      v8f racc;
#pragma unroll
      for (int j = 0; j < 8; ++j) racc[j] = brv;
#pragma unroll
      for (int kb = 0; kb < 4; ++kb)
        racc = wmma16(afrag_lds(xh, XHc, kb * 32, lane), wra[kb], racc);
#pragma unroll
      for (int j = 0; j < 8; ++j)
        r_lds[(mrow + j) * Hc + ncol] = sigmoidf_(racc[j]);
    }
    __syncthreads();

    // Phase B: overwrite concat tail with r * h (8B store per thread).
    {
      const float* rp = r_lds + srow * Hc + sc4;
      const float* hp = h_lds + srow * Hc + sc4;
      v4h v;
#pragma unroll
      for (int j = 0; j < 4; ++j) v[j] = (_Float16)(rp[j] * hp[j]);
      *(v4h*)(xh + srow * XHc + Ec + sc4) = v;
    }
    __syncthreads();

    // Phase C: hbar = tanh(xhr @ Wh^T + bh); h = h + z*(hbar - h).
    if (wave < 4) {
      v8f hacc;
#pragma unroll
      for (int j = 0; j < 8; ++j) hacc[j] = bhv;
#pragma unroll
      for (int kb = 0; kb < 4; ++kb)
        hacc = wmma16(afrag_lds(xh, XHc, kb * 32, lane), wha[kb], hacc);
#pragma unroll
      for (int j = 0; j < 8; ++j) {
        int o = (mrow + j) * Hc + ncol;
        float hp = h_lds[o];
        h_lds[o] = hp + zacc[j] * (tanhf_(hacc[j]) - hp);
      }
    }
    __syncthreads();

    // Coalesced f16 store of hidden[b0..b0+15, t, :] (8B per thread).
    {
      const float* hp = h_lds + srow * Hc + sc4;
      v4h hv;
#pragma unroll
      for (int j = 0; j < 4; ++j) hv[j] = (_Float16)hp[j];
      *(v4h*)(hidden + ((size_t)(b0 + srow) * Tc + t) * Hc + sc4) = hv;
    }
    // h_lds next written in Phase C (3 barriers away); xh next written after
    // the Phase-C barrier: no extra barrier needed here.
  }
}

// ---------------------------------------------------------------------------
// Kernel 2: logits = hidden @ Wlm^T + blm.  [131072 x 64] x [64 x 128].
// One block = one 16-row M tile; 8 waves = 8 N-tiles of 16 (V=128).
// ---------------------------------------------------------------------------
__global__ __launch_bounds__(256, 2) void lmhead_kernel(
    const _Float16* __restrict__ hidden, const float* __restrict__ Wlm,
    const float* __restrict__ blm, float* __restrict__ out)
{
  __shared__ __align__(16) _Float16 atile[16 * Hc];   // 2 KB contiguous A tile

  const int tid  = threadIdx.x;
  const int lane = tid & 31;
  const int wave = tid >> 5;                       // N-tile 0..7
  const size_t m0 = (size_t)blockIdx.x * 16;       // flattened [B*T] row tile

  *(v4h*)(atile + tid * 4) = *(const v4h*)(hidden + m0 * Hc + tid * 4);
  __syncthreads();

  v16h a0 = afrag_lds(atile, Hc, 0, lane);
  v16h a1 = afrag_lds(atile, Hc, 32, lane);
  v16h b0 = bfrag_f32(Wlm, Hc, wave * 16, 0, lane);
  v16h b1 = bfrag_f32(Wlm, Hc, wave * 16, 32, lane);

  const int ncol = wave * 16 + (lane & 15);
  const int mrow = (lane >> 4) << 3;
  v8f acc;
  const float bias = blm[ncol];
#pragma unroll
  for (int j = 0; j < 8; ++j) acc[j] = bias;
  acc = wmma16(a0, b0, acc);
  acc = wmma16(a1, b1, acc);

#pragma unroll
  for (int j = 0; j < 8; ++j)
    out[(m0 + mrow + j) * Vc + ncol] = acc[j];
}

// ---------------------------------------------------------------------------
extern "C" void kernel_launch(void* const* d_in, const int* in_sizes, int n_in,
                              void* d_out, int out_size, void* d_ws, size_t ws_size,
                              hipStream_t stream) {
  const int*   idx   = (const int*)  d_in[0];
  const float* start = (const float*)d_in[1];
  const float* wte   = (const float*)d_in[2];
  const float* Wz    = (const float*)d_in[3];
  const float* bz    = (const float*)d_in[4];
  const float* Wr    = (const float*)d_in[5];
  const float* br    = (const float*)d_in[6];
  const float* Wh    = (const float*)d_in[7];
  const float* bh    = (const float*)d_in[8];
  const float* Wlm   = (const float*)d_in[9];
  const float* blm   = (const float*)d_in[10];
  float* out = (float*)d_out;

  _Float16* hidden = (_Float16*)d_ws;   // B*T*H f16 = 16 MiB scratch

  gru_scan_kernel<<<Bc / 16, 256, 0, stream>>>(idx, start, wte, Wz, bz, Wr, br,
                                               Wh, bh, hidden);
  lmhead_kernel<<<(Bc * Tc) / 16, 256, 0, stream>>>(hidden, Wlm, blm, out);
}
